// BoundaryLoss_15805479649569
// MI455X (gfx1250) — 8-bit matrix operand order repaired and checked by reading, compiled, not run on hardware
//
#include <hip/hip_runtime.h>


#define BATCH 16
#define NCLS  4
#define IMH   512
#define IMW   512
#define NPIX  (IMH * IMW)
#define RAD   5
#define TILE  32
#define HALO  (TILE + 2 * RAD)
#define MSTR  48
#define PLANE (HALO * MSTR)
#define KVAL  121
#define NRING 14

typedef int v8i __attribute__((ext_vector_type(8)));
typedef unsigned int u32x4 __attribute__((ext_vector_type(4)));
typedef int i32x4 __attribute__((ext_vector_type(4)));
typedef int i32x8 __attribute__((ext_vector_type(8)));

static __device__ constexpr int ringClass(int k) {
  if (k >= KVAL) return 15;
  const int dy = k / 11 - RAD, dx = k % 11 - RAD;
  const int d2 = dy * dy + dx * dx;
  const int l[NRING] = {0, 1, 2, 4, 5, 8, 9, 10, 13, 16, 17, 18, 20, 25};
  for (int i = 0; i < NRING; ++i)
    if (l[i] == d2) return i;
  return 15;
}

static __device__ inline v8i wmma_iu8(v8i a, v8i b, v8i c) {
  return __builtin_amdgcn_wmma_i32_16x16x64_iu8(false, a, false, b, c, false, false);
}

template <int HALF, int STEP>
static __device__ inline v8i buildRingA(int ring) {
  v8i aa;
#pragma unroll
  for (int v = 0; v < 8; ++v) {
    unsigned w = 0;
#pragma unroll
    for (int j = 0; j < 4; ++j) {
      const int k = ((v >> 1) << 4) + (HALF << 3) + ((v & 1) << 2) + (STEP << 6) + j;
      w |= ((ringClass(k) == ring) ? 1u : 0u) << (8 * j);
    }
    aa[v] = (int)w;
  }
  return aa;
}

template <int KHALF, int STEP>
static __device__ inline v8i buildBim(const unsigned char* mb, int center) {
  v8i bb;
#pragma unroll
  for (int v = 0; v < 8; ++v) {
    unsigned w = 0;
#pragma unroll
    for (int j = 0; j < 4; ++j) {
      const int K = ((v >> 1) << 4) + (KHALF << 3) + ((v & 1) << 2) + (STEP << 6) + j;
      if (K < KVAL) {
        const int dy = K / 11 - RAD, dx = K % 11 - RAD;
        w |= (unsigned)mb[center + (dy * MSTR + dx)] << (8 * j);
      }
    }
    bb[v] = (int)w;
  }
  return bb;
}

__global__ void bl_zero_ws(unsigned* ws) {
  if (threadIdx.x < 160) ws[threadIdx.x] = 0u;
}

__global__ __launch_bounds__(256) void bl_flags(const float* __restrict__ pred,
                                                const int* __restrict__ tgt,
                                                float* __restrict__ sumP,
                                                unsigned* __restrict__ cntT) {
  const int bc = blockIdx.x;
  const int b = bc >> 2, c = bc & 3;
  const int n = NPIX / 32;
  const int begin = blockIdx.y * n;
  const size_t baseT = (size_t)b * NPIX;
  const size_t baseP = (size_t)b * NCLS * NPIX;

  float s = 0.f;
  unsigned cnt = 0;
  for (int i = threadIdx.x; i < n; i += 256) {
    const int idx = begin + i;
    cnt += (tgt[baseT + idx] == c) ? 1u : 0u;
    const float x0 = pred[baseP + 0 * NPIX + idx];
    const float x1 = pred[baseP + 1 * NPIX + idx];
    const float x2 = pred[baseP + 2 * NPIX + idx];
    const float x3 = pred[baseP + 3 * NPIX + idx];
    __builtin_prefetch(&pred[baseP + idx + 4096], 0, 0);
    const float mx = fmaxf(fmaxf(x0, x1), fmaxf(x2, x3));
    const float e0 = expf(x0 - mx), e1 = expf(x1 - mx);
    const float e2 = expf(x2 - mx), e3 = expf(x3 - mx);
    const float den = e0 + e1 + e2 + e3;
    s += ((c == 0) ? e0 : (c == 1) ? e1 : (c == 2) ? e2 : e3) / den;
  }
  for (int sh = 16; sh >= 1; sh >>= 1) {
    s += __shfl_xor(s, sh, 32);
    cnt += __shfl_xor(cnt, sh, 32);
  }
  if ((threadIdx.x & 31) == 0) {
    atomicAdd(&sumP[bc], s);
    atomicAdd(&cntT[bc], cnt);
  }
}

__global__ __launch_bounds__(256) void bl_edt_main(const float* __restrict__ pred,
                                                   const int* __restrict__ tgt,
                                                   const float* __restrict__ sumP,
                                                   const unsigned* __restrict__ cntT,
                                                   float* __restrict__ gLoss) {
  __shared__ int rawT[HALO * HALO];
  __shared__ unsigned char msk[NCLS][4][PLANE];
  __shared__ unsigned anyMask;
  __shared__ float ldsAcc;

  const int tid = threadIdx.x;
  const int lane = tid & 31;
  const int wave = tid >> 5;
  const int half = lane >> 4;
  const int m = lane & 15;
  const int b = blockIdx.z;
  const int tx0 = blockIdx.x * TILE, ty0 = blockIdx.y * TILE;
  const int gx0 = tx0 - RAD, gy0 = ty0 - RAD;
  const int cx0 = (gx0 < 0) ? 0 : gx0;
  const int cy0 = (gy0 < 0) ? 0 : gy0;
  const int cx1 = (tx0 + TILE + RAD < IMW) ? (tx0 + TILE + RAD) : IMW;
  const int cy1 = (ty0 + TILE + RAD < IMH) ? (ty0 + TILE + RAD) : IMH;
  const int tw = cx1 - cx0, th = cy1 - cy0;

  if (tid == 0) {
    ldsAcc = 0.f;
    anyMask = 0u;
  }

  if (wave == 0) {
    const unsigned ldsBase = (unsigned)(size_t)(&rawT[0]);
    const unsigned long long ga =
        (unsigned long long)(size_t)(const void*)(tgt + ((size_t)b * IMH + cy0) * IMW + cx0);
    u32x4 g0 = {1u,
                ldsBase,
                (unsigned)ga,
                (unsigned)((ga >> 32) & 0x1FFFFFFull) | (2u << 30)};
    i32x8 g1 = {(int)(2u << 16),
                (int)((unsigned)IMW << 16),
                (int)((unsigned)IMH << 16),
                (int)((unsigned)tw << 16),
                (int)(unsigned)th,
                (int)IMW,
                0, 0};
    i32x4 g2 = {0, 0, 0, 0};
    i32x4 g3 = {0, 0, 0, 0};
#if defined(__clang_major__) && (__clang_major__ >= 23)
    i32x8 g4 = {0, 0, 0, 0, 0, 0, 0, 0};
    __builtin_amdgcn_tensor_load_to_lds(g0, g1, g2, g3, g4, 0);
#else
    __builtin_amdgcn_tensor_load_to_lds(g0, g1, g2, g3, 0);
#endif
    __builtin_amdgcn_s_wait_tensorcnt(0);
  }

  {
    unsigned* mz = (unsigned*)&msk[0][0][0];
    for (int i = tid; i < NCLS * 4 * PLANE / 4; i += 256) mz[i] = 0u;
  }
  __syncthreads();

  unsigned lbits = 0;
  for (int p = tid; p < HALO * HALO; p += 256) {
    const int hy = p / HALO, hx = p - hy * HALO;
    const int gx = gx0 + hx, gy = gy0 + hy;
    if ((gx >= 0) & (gx < IMW) & (gy >= 0) & (gy < IMH)) {
      const int idx = hy * MSTR + hx;
      const int t = rawT[(gy - cy0) * tw + (gx - cx0)];
      msk[t][0][idx] = 1;
      lbits |= 1u << (t * 4);
#pragma unroll
      for (int cc = 0; cc < 4; ++cc)
        if (cc != t) { msk[cc][1][idx] = 1; lbits |= 1u << (cc * 4 + 1); }

      const size_t baseP = ((size_t)b * NCLS * IMH + gy) * IMW + gx;
      float x[4];
#pragma unroll
      for (int ch = 0; ch < 4; ++ch) x[ch] = pred[baseP + (size_t)ch * NPIX];
      const float mx = fmaxf(fmaxf(x[0], x[1]), fmaxf(x[2], x[3]));
      float e[4], den = 0.f;
#pragma unroll
      for (int ch = 0; ch < 4; ++ch) { e[ch] = expf(x[ch] - mx); den += e[ch]; }
#pragma unroll
      for (int cc = 0; cc < 4; ++cc) {
        const float pc = e[cc] / den;
        if (pc == 1.0f) { msk[cc][2][idx] = 1; lbits |= 1u << (cc * 4 + 2); }
        if (pc == 0.0f) { msk[cc][3][idx] = 1; lbits |= 1u << (cc * 4 + 3); }
      }
    }
  }
  if (lbits) atomicOr(&anyMask, lbits);
  __syncthreads();

  v8i Aop0, Aop1;
  if (half) { Aop0 = buildRingA<1, 0>(m); Aop1 = buildRingA<1, 1>(m); }
  else      { Aop0 = buildRingA<0, 0>(m); Aop1 = buildRingA<0, 1>(m); }
  const float RL[8] = {0.0f, 1.0f, 1.41421356f, 2.0f,
                       2.23606798f, 2.82842712f, 3.0f, 3.16227766f};
  const float RH[8] = {3.60555128f, 4.0f, 4.12310563f, 4.24264069f,
                       4.47213595f, 5.0f, 5.0f, 5.0f};
  float rD[8];
#pragma unroll
  for (int v = 0; v < 8; ++v) rD[v] = half ? RH[v] : RL[v];

  const unsigned anyBits = anyMask;
  float laneAcc = 0.f;

  for (int cc = 0; cc < NCLS; ++cc) {
    const int bc = b * NCLS + cc;
    const bool tE = (cntT[bc] == 0u);
    const bool pE = (sumP[bc] == 0.0f);

    for (int g = wave; g < 64; g += 8) {
      const int py = g >> 1;
      const int px = ((g & 1) << 4) + m;
      const int center = (py + RAD) * MSTR + (px + RAD);

      float dvv[4];
#pragma unroll
      for (int t = 0; t < 4; ++t) {
        if (anyBits & (1u << (cc * 4 + t))) {
          const unsigned char* mb = &msk[cc][t][0];
          v8i b0, b1;
          if (half) { b0 = buildBim<1, 0>(mb, center); b1 = buildBim<1, 1>(mb, center); }
          else      { b0 = buildBim<0, 0>(mb, center); b1 = buildBim<0, 1>(mb, center); }
          v8i acc = {};
          acc = wmma_iu8(Aop0, b0, acc);
          acc = wmma_iu8(Aop1, b1, acc);
          float cand = 5.0f;
#pragma unroll
          for (int v = 0; v < 8; ++v)
            cand = fminf(cand, (acc[v] != 0) ? rD[v] : 5.0f);
          dvv[t] = fminf(cand, __shfl_xor(cand, 16, 32));
        } else {
          dvv[t] = 5.0f;
        }
      }
      const float st = tE ? 3.0f
                          : fminf(5.f, fmaxf(-5.f, dvv[0] - dvv[1])) / 5.0f;
      const float sp = pE ? 3.0f
                          : fminf(5.f, fmaxf(-5.f, dvv[2] - dvv[3])) / 5.0f;
      laneAcc += fabsf(sp - st);
    }
  }

  for (int sh = 16; sh >= 1; sh >>= 1) laneAcc += __shfl_xor(laneAcc, sh, 32);
  if (lane == 0) atomicAdd(&ldsAcc, laneAcc);
  __syncthreads();
  if (tid == 0) atomicAdd(gLoss, ldsAcc);
}

__global__ void bl_finalize(const float* __restrict__ gLoss, float* __restrict__ out) {
  if (threadIdx.x == 0) out[0] = gLoss[0] / 33554432.0f;
}

extern "C" void kernel_launch(void* const* d_in, const int* in_sizes, int n_in,
                              void* d_out, int out_size, void* d_ws, size_t ws_size,
                              hipStream_t stream) {
  const float* pred = (const float*)d_in[0];
  const int* tgt = (const int*)d_in[1];

  float* sumP = (float*)d_ws;
  unsigned* cntT = (unsigned*)((char*)d_ws + 256);
  float* gLoss = (float*)((char*)d_ws + 512);

  bl_zero_ws<<<1, 256, 0, stream>>>((unsigned*)d_ws);
  bl_flags<<<dim3(BATCH * NCLS, 32), 256, 0, stream>>>(pred, tgt, sumP, cntT);
  bl_edt_main<<<dim3(IMW / TILE, IMH / TILE, BATCH), 256, 0, stream>>>(
      pred, tgt, sumP, cntT, gLoss);
  bl_finalize<<<1, 32, 0, stream>>>(gLoss, (float*)d_out);
}
